// MultiHeadAttention_42339787604363
// MI455X (gfx1250) — matrix operand order repaired and checked by reading, not compiled, not run on hardware
//
#include <hip/hip_runtime.h>
#include <hip/hip_bf16.h>

typedef _Float16 f16;
typedef __attribute__((ext_vector_type(16))) _Float16 v16h;
typedef __attribute__((ext_vector_type(8)))  _Float16 v8h;
typedef __attribute__((ext_vector_type(8)))  float    v8f;

#define HIDDEN   768
#define SEQ      4096
#define HEADS    12
#define HEAD_DIM 64

__device__ __forceinline__ v8f wmma16x16x32(v16h a, v16h b, v8f c) {
  return __builtin_amdgcn_wmma_f32_16x16x32_f16(false, a, false, b, (short)0, c,
                                                false, false);
}

__device__ __forceinline__ v8f vzero8() {
  v8f z;
#pragma unroll
  for (int i = 0; i < 8; ++i) z[i] = 0.0f;
  return z;
}

// Build a v16h A/B fragment from two contiguous 16-byte LDS chunks.
__device__ __forceinline__ v16h frag_from2(const f16* p0, const f16* p1) {
  v8h lo = *(const v8h*)p0;
  v8h hi = *(const v8h*)p1;
  v16h r;
#pragma unroll
  for (int e = 0; e < 8; ++e) {
    r[e] = lo[e];
    r[8 + e] = hi[e];
  }
  return r;
}

// ---- 16-lane butterfly reductions via DPP16 (no LDS, no dscnt waits) ------
// quad_perm[1,0,3,2]=0xB1 (xor1), quad_perm[2,3,0,1]=0x4E (xor2),
// row_half_mirror=0x141 (combine quads), row_mirror=0x140 (combine halves).
template <int CTRL>
__device__ __forceinline__ float dpp_mov(float v) {
  return __int_as_float(__builtin_amdgcn_update_dpp(
      0, __float_as_int(v), CTRL, 0xf, 0xf, true));
}
__device__ __forceinline__ float red16_max(float v) {
  v = fmaxf(v, dpp_mov<0xB1>(v));
  v = fmaxf(v, dpp_mov<0x4E>(v));
  v = fmaxf(v, dpp_mov<0x141>(v));
  v = fmaxf(v, dpp_mov<0x140>(v));
  return v;
}
__device__ __forceinline__ float red16_sum(float v) {
  v = v + dpp_mov<0xB1>(v);
  v = v + dpp_mov<0x4E>(v);
  v = v + dpp_mov<0x141>(v);
  v = v + dpp_mov<0x140>(v);
  return v;
}

// 8x GLOBAL_LOAD_ASYNC_TO_LDS_B128: one 64B K-row half + one 64B V-row half.
#define ISSUE_ASYNC_KV(lk, lv, gk, gv)                                  \
  asm volatile(                                                         \
      "global_load_async_to_lds_b128 %0, %2, off\n\t"                   \
      "global_load_async_to_lds_b128 %0, %2, off offset:16\n\t"         \
      "global_load_async_to_lds_b128 %0, %2, off offset:32\n\t"         \
      "global_load_async_to_lds_b128 %0, %2, off offset:48\n\t"         \
      "global_load_async_to_lds_b128 %1, %3, off\n\t"                   \
      "global_load_async_to_lds_b128 %1, %3, off offset:16\n\t"         \
      "global_load_async_to_lds_b128 %1, %3, off offset:32\n\t"         \
      "global_load_async_to_lds_b128 %1, %3, off offset:48"             \
      :                                                                 \
      : "v"(lk), "v"(lv), "v"(gk), "v"(gv)                              \
      : "memory")

// ---------------------------------------------------------------------------
// Y = (X @ W + bias) * oscale ; optionally stored transposed (Y[col][row]).
// BM=128, BN=128, BK=32; 256 threads = 8 waves (2x4), 8 WMMAs/wave/K-step.
// ---------------------------------------------------------------------------
template <typename TIN, bool OUT_F16, bool TRANS_OUT>
__global__ __launch_bounds__(256) void gemm_bias_kernel(
    const TIN* __restrict__ X, const float* __restrict__ W,
    const float* __restrict__ bias, void* __restrict__ Yv, int M, int N, int K,
    float oscale) {
  constexpr int BM = 128, BN = 128, BK = 32;
  constexpr int ASTR = BK + 8;
  constexpr int TSTR = BK + 8;
  __shared__ __align__(16) f16 As[BM * ASTR];
  __shared__ __align__(16) f16 BsT[BN * TSTR];

  const int mb = blockIdx.x * BM;
  const int nb = blockIdx.y * BN;
  const int t = threadIdx.x;
  const int lane = t & 31;
  const int wid = t >> 5;
  const int wm = (wid >> 2) * 64;
  const int wn = (wid & 3) * 32;

  const int ar = t >> 1, ac = (t & 1) * 16;
  const TIN* asrc = X + (size_t)(mb + ar) * K + ac;
  const int bn = t & 127, bk0 = (t >> 7) * 16;
  const float* bsrc = W + (size_t)bk0 * N + nb + bn;

  v8f acc[4][2];
#pragma unroll
  for (int i = 0; i < 4; ++i)
#pragma unroll
    for (int j = 0; j < 2; ++j) acc[i][j] = vzero8();

  for (int kb = 0; kb < K; kb += BK) {
    {  // A tile -> LDS row-major f16
      const TIN* src = asrc + kb;
      f16* dst = As + ar * ASTR + ac;
      v8h w0, w1;
#pragma unroll
      for (int e = 0; e < 8; ++e) {
        w0[e] = (f16)src[e];
        w1[e] = (f16)src[8 + e];
      }
      *(v8h*)dst = w0;
      *(v8h*)(dst + 8) = w1;
    }
    {  // B tile -> LDS transposed: BsT[n][k]
      const float* src = bsrc + (size_t)kb * N;
      v8h w0, w1;
#pragma unroll
      for (int e = 0; e < 8; ++e) {
        w0[e] = (f16)src[(size_t)e * N];
        w1[e] = (f16)src[(size_t)(8 + e) * N];
      }
      f16* dst = BsT + bn * TSTR + bk0;
      *(v8h*)dst = w0;
      *(v8h*)(dst + 8) = w1;
    }
    if (kb + BK < K) {  // -> global_prefetch_b8
      __builtin_prefetch(asrc + kb + BK, 0, 1);
      __builtin_prefetch(bsrc + (size_t)(kb + BK) * N, 0, 1);
    }
    __syncthreads();

    v16h bfrag[2];
#pragma unroll
    for (int j = 0; j < 2; ++j) {
      const f16* p = BsT + (wn + j * 16 + (lane & 15)) * TSTR + (lane >> 4) * 16;
      bfrag[j] = frag_from2(p, p + 8);
    }
#pragma unroll
    for (int i = 0; i < 4; ++i) {
      const f16* p = As + (wm + i * 16 + (lane & 15)) * ASTR + (lane >> 4) * 16;
      v16h a = frag_from2(p, p + 8);
#pragma unroll
      for (int j = 0; j < 2; ++j) acc[i][j] = wmma16x16x32(a, bfrag[j], acc[i][j]);
    }
    __syncthreads();
  }

#pragma unroll
  for (int i = 0; i < 4; ++i) {
#pragma unroll
    for (int j = 0; j < 2; ++j) {
      const int col = nb + wn + j * 16 + (lane & 15);
      const float bb = bias[col];
#pragma unroll
      for (int r = 0; r < 8; ++r) {
        const int row = mb + wm + i * 16 + (lane >> 4) * 8 + r;
        const float v = (acc[i][j][r] + bb) * oscale;
        const size_t idx = TRANS_OUT ? ((size_t)col * M + row) : ((size_t)row * N + col);
        if (OUT_F16)
          ((f16*)Yv)[idx] = (f16)v;
        else
          ((float*)Yv)[idx] = v;
      }
    }
  }
}

// ---------------------------------------------------------------------------
// Flash attention: grid (SEQ/64, HEADS), 128 threads = 4 waves.
// Q pre-scaled by 1/sqrt(d); V provided transposed ([head*d][seq]).
// Double-buffered GLOBAL_LOAD_ASYNC_TO_LDS_B128 staging; DPP16 softmax
// reductions (no LDS round trips).
// ---------------------------------------------------------------------------
__global__ __launch_bounds__(128) void flash_attn_kernel(
    const f16* __restrict__ Qg, const f16* __restrict__ Kg,
    const f16* __restrict__ Vt, f16* __restrict__ Og) {
  constexpr int BKV = 64;
  constexpr int KSTR = HEAD_DIM + 8;  // 72 f16
  constexpr int NB = SEQ / BKV;
  __shared__ __align__(16) f16 Ks[2][BKV * KSTR];
  __shared__ __align__(16) f16 Vs[2][BKV * KSTR];  // V^T tile: [d][key]
  __shared__ __align__(16) f16 Ps[4 * 16 * KSTR];

  const int head = blockIdx.y;
  const int qb = blockIdx.x * 64;
  const int t = threadIdx.x;
  const int lane = t & 31;
  const int wid = t >> 5;

  v16h aq[2];
  {
    const int m = qb + wid * 16 + (lane & 15);
    const f16* qp = Qg + (size_t)m * HIDDEN + head * HEAD_DIM + (lane >> 4) * 16;
#pragma unroll
    for (int c = 0; c < 2; ++c) aq[c] = frag_from2(qp + c * 32, qp + c * 32 + 8);
  }

  const int srow = t >> 1, shalf = t & 1;
  const unsigned lds_k0 = (unsigned)(uintptr_t)(&Ks[0][srow * KSTR + shalf * 32]);
  const unsigned lds_k1 = (unsigned)(uintptr_t)(&Ks[1][srow * KSTR + shalf * 32]);
  const unsigned lds_v0 = (unsigned)(uintptr_t)(&Vs[0][srow * KSTR + shalf * 32]);
  const unsigned lds_v1 = (unsigned)(uintptr_t)(&Vs[1][srow * KSTR + shalf * 32]);
  const unsigned long long gk0 = (unsigned long long)(uintptr_t)(
      Kg + (size_t)srow * HIDDEN + head * HEAD_DIM + shalf * 32);
  const unsigned long long gv0 = (unsigned long long)(uintptr_t)(
      Vt + (size_t)(head * HEAD_DIM + srow) * SEQ + shalf * 32);
  const unsigned long long kstep = (unsigned long long)BKV * HIDDEN * sizeof(f16);
  const unsigned long long vstep = (unsigned long long)BKV * sizeof(f16);

  float mrow[8], lrow[8];
  v8f o[4];
#pragma unroll
  for (int r = 0; r < 8; ++r) {
    mrow[r] = -1e30f;
    lrow[r] = 0.0f;
  }
#pragma unroll
  for (int dt = 0; dt < 4; ++dt) o[dt] = vzero8();

  ISSUE_ASYNC_KV(lds_k0, lds_v0, gk0, gv0);  // stage block 0 -> buffer 0

  for (int i = 0; i < NB; ++i) {
    const int nxt = (i + 1 < NB) ? i + 1 : i;
    const unsigned lk = ((i + 1) & 1) ? lds_k1 : lds_k0;
    const unsigned lv = ((i + 1) & 1) ? lds_v1 : lds_v0;
    ISSUE_ASYNC_KV(lk, lv, gk0 + (unsigned long long)nxt * kstep,
                   gv0 + (unsigned long long)nxt * vstep);
    // ASYNCcnt in-order: <=8 outstanding => current block's 8 loads landed
    asm volatile("s_wait_asynccnt 0x8" ::: "memory");
    __syncthreads();

    const f16* Kc = Ks[i & 1];
    const f16* Vc = Vs[i & 1];

    // S = Q @ K^T  (4 key tiles x 2 d-chunks)
    v8f s[4];
#pragma unroll
    for (int kt = 0; kt < 4; ++kt) s[kt] = vzero8();
#pragma unroll
    for (int kt = 0; kt < 4; ++kt) {
#pragma unroll
      for (int c = 0; c < 2; ++c) {
        const f16* p = Kc + (kt * 16 + (lane & 15)) * KSTR + c * 32 + (lane >> 4) * 16;
        s[kt] = wmma16x16x32(aq[c], frag_from2(p, p + 8), s[kt]);
      }
    }

    // online softmax (lanes 0-15 own rows 0..7, lanes 16-31 own rows 8..15)
    float mnew[8], alpha[8];
#pragma unroll
    for (int r = 0; r < 8; ++r) {
      float v = fmaxf(fmaxf(s[0][r], s[1][r]), fmaxf(s[2][r], s[3][r]));
      v = red16_max(v);
      mnew[r] = fmaxf(mrow[r], v);
      alpha[r] = __expf(mrow[r] - mnew[r]);
      mrow[r] = mnew[r];
    }
#pragma unroll
    for (int r = 0; r < 8; ++r) {
      float ps = 0.0f;
#pragma unroll
      for (int kt = 0; kt < 4; ++kt) {
        const float e = __expf(s[kt][r] - mnew[r]);
        s[kt][r] = e;
        ps += e;
      }
      ps = red16_sum(ps);
      lrow[r] = lrow[r] * alpha[r] + ps;
#pragma unroll
      for (int dt = 0; dt < 4; ++dt) o[dt][r] *= alpha[r];
    }

    // restage P: C layout -> per-wave LDS (row-major 16x64 f16)
    {
      f16* pw = Ps + wid * 16 * KSTR;
      const int rowbase = (lane >> 4) * 8;
#pragma unroll
      for (int kt = 0; kt < 4; ++kt) {
        const int col = kt * 16 + (lane & 15);
#pragma unroll
        for (int r = 0; r < 8; ++r) pw[(rowbase + r) * KSTR + col] = (f16)s[kt][r];
      }
    }

    // O += P @ V : B fragment n=d, k=key -> contiguous rows of V^T tile
#pragma unroll
    for (int c = 0; c < 2; ++c) {
      const f16* pw = Ps + wid * 16 * KSTR + (lane & 15) * KSTR + c * 32 + (lane >> 4) * 16;
      v16h ap = frag_from2(pw, pw + 8);
#pragma unroll
      for (int dt = 0; dt < 4; ++dt) {
        const f16* p = Vc + (dt * 16 + (lane & 15)) * KSTR + c * 32 + (lane >> 4) * 16;
        o[dt] = wmma16x16x32(ap, frag_from2(p, p + 8), o[dt]);
      }
    }
    __syncthreads();
  }

#pragma unroll
  for (int r = 0; r < 8; ++r) lrow[r] = 1.0f / lrow[r];
#pragma unroll
  for (int dt = 0; dt < 4; ++dt) {
    const int col = head * HEAD_DIM + dt * 16 + (lane & 15);
#pragma unroll
    for (int r = 0; r < 8; ++r) {
      const int row = qb + wid * 16 + (lane >> 4) * 8 + r;
      Og[(size_t)row * HIDDEN + col] = (f16)(o[dt][r] * lrow[r]);
    }
  }
}

// ---------------------------------------------------------------------------
extern "C" void kernel_launch(void* const* d_in, const int* in_sizes, int n_in,
                              void* d_out, int out_size, void* d_ws, size_t ws_size,
                              hipStream_t stream) {
  (void)in_sizes; (void)n_in; (void)out_size; (void)ws_size;
  const float* x  = (const float*)d_in[0];
  const float* Wq = (const float*)d_in[1];
  const float* bq = (const float*)d_in[2];
  const float* Wk = (const float*)d_in[3];
  const float* bk = (const float*)d_in[4];
  const float* Wv = (const float*)d_in[5];
  const float* bv = (const float*)d_in[6];
  const float* Wo = (const float*)d_in[7];
  const float* bo = (const float*)d_in[8];

  const size_t elems = (size_t)SEQ * HIDDEN;
  f16* Qb  = (f16*)d_ws;
  f16* Kb  = Qb + elems;
  f16* Vtb = Kb + elems;  // V stored transposed: [head*d][seq]
  f16* Ob  = Vtb + elems;

  const float qscale = 0.125f;  // 1/sqrt(HEAD_DIM) folded into Q projection

  dim3 ggrid(SEQ / 128, HIDDEN / 128);
  dim3 gblk(256);
  gemm_bias_kernel<float, true, false><<<ggrid, gblk, 0, stream>>>(
      x, Wq, bq, Qb, SEQ, HIDDEN, HIDDEN, qscale);
  gemm_bias_kernel<float, true, false><<<ggrid, gblk, 0, stream>>>(
      x, Wk, bk, Kb, SEQ, HIDDEN, HIDDEN, 1.0f);
  gemm_bias_kernel<float, true, true><<<ggrid, gblk, 0, stream>>>(
      x, Wv, bv, Vtb, SEQ, HIDDEN, HIDDEN, 1.0f);

  dim3 agrid(SEQ / 64, HEADS);
  dim3 ablk(128);
  flash_attn_kernel<<<agrid, ablk, 0, stream>>>(Qb, Kb, Vtb, Ob);

  gemm_bias_kernel<f16, false, false><<<ggrid, gblk, 0, stream>>>(
      Ob, Wo, bo, (float*)d_out, SEQ, HIDDEN, HIDDEN, 1.0f);
}
